// MultiHeadLatentAttention_6107443495280
// MI455X (gfx1250) — matrix operand order repaired and checked by reading, not compiled, not run on hardware
//
#include <hip/hip_runtime.h>

// ---------------------------------------------------------------------------
// CDNA5 bf16 WMMA fragment plumbing
// ---------------------------------------------------------------------------
typedef __attribute__((ext_vector_type(16))) __bf16 v16bf;
typedef __attribute__((ext_vector_type(8)))  float  v8f;

union ABfrag { v16bf v; unsigned int u[8]; };

__device__ __forceinline__ unsigned short f2bf(float f) {
    unsigned int u = __float_as_uint(f);
    u += 0x7FFFu + ((u >> 16) & 1u);          // round to nearest even
    return (unsigned short)(u >> 16);
}
__device__ __forceinline__ unsigned int pack2(float a, float b) {
    return (unsigned int)f2bf(a) | ((unsigned int)f2bf(b) << 16);
}
__device__ __forceinline__ v8f wmma_bf16(const ABfrag& a, const ABfrag& b, v8f c) {
    return __builtin_amdgcn_wmma_f32_16x16x32_bf16(
        /*neg_a=*/false, a.v, /*neg_b=*/false, b.v,
        /*c_mod=*/(short)0, c, /*reuse_a=*/false, /*reuse_b=*/false);
}

// ---------------------------------------------------------------------------
// GEMM: C[M,N] = A[M,K] (f32, row-major) * B[N,K]^T (f32, row-major)
// bf16 WMMA, f32 accumulate.  Block tile 128x128, BK=32, 256 threads = 8
// wave32.  Wave (wm,wn) in 2x4 grid computes 64(M) x 32(N): 8 WMMAs/k-step.
// Double-buffered LDS: stage tile k+1 while the matrix pipe chews tile k.
// ---------------------------------------------------------------------------
#define BM 128
#define BN 128
#define BK 32
#define TILE_DW (BM * BK / 2)           // bf16 pairs: 128 rows x 16 dwords

// f32 -> bf16 convert + store one 1x16 strip of a staging tile into LDS
__device__ __forceinline__ void stage16(const float* __restrict__ g,
                                        unsigned int* __restrict__ dst) {
    const float4* gv = (const float4*)g;
    float4 f0 = gv[0], f1 = gv[1], f2 = gv[2], f3 = gv[3];
    dst[0] = pack2(f0.x, f0.y); dst[1] = pack2(f0.z, f0.w);
    dst[2] = pack2(f1.x, f1.y); dst[3] = pack2(f1.z, f1.w);
    dst[4] = pack2(f2.x, f2.y); dst[5] = pack2(f2.z, f2.w);
    dst[6] = pack2(f3.x, f3.y); dst[7] = pack2(f3.z, f3.w);
}

__global__ __launch_bounds__(256) void gemm_bf16(const float* __restrict__ A,
                                                 const float* __restrict__ B,
                                                 float* __restrict__ C,
                                                 int M, int N, int K) {
    __shared__ unsigned int lAs[2][TILE_DW];
    __shared__ unsigned int lBs[2][TILE_DW];

    const int bm = blockIdx.y * BM;
    const int bn = blockIdx.x * BN;
    const int tid  = threadIdx.x;
    const int wave = tid >> 5;
    const int lane = tid & 31;
    const int nn = lane & 15;
    const int hi = lane >> 4;
    const int wm = wave >> 2;       // 0..1 : 64-row strip
    const int wn = wave & 3;        // 0..3 : 32-col strip

    v8f acc[4][2];
    v8f zero = {};
    #pragma unroll
    for (int mt = 0; mt < 4; ++mt)
        #pragma unroll
        for (int nt = 0; nt < 2; ++nt) acc[mt][nt] = zero;

    const int r  = tid >> 1;            // staging row within tile (0..127)
    const int cc = (tid & 1) * 16;      // staging col offset (0 or 16)
    const float* gA = A + (size_t)(bm + r) * K + cc;
    const float* gB = B + (size_t)(bn + r) * K + cc;
    const int dstOff = r * 16 + (cc >> 1);

    // prologue: stage tile 0 into buffer 0
    stage16(gA, lAs[0] + dstOff);
    stage16(gB, lBs[0] + dstOff);

    int buf = 0;
    for (int k0 = 0; k0 < K; k0 += BK) {
        __syncthreads();    // staging of buf complete; buf^1 free to overwrite

        if (k0 + BK < K) {
            if (k0 + 2 * BK < K) {      // WGP-scope hint for the tile after next
                __builtin_prefetch(gA + k0 + 2 * BK, 0, 3);
                __builtin_prefetch(gB + k0 + 2 * BK, 0, 3);
            }
            // stage tile k+1 into the other buffer while we compute tile k
            stage16(gA + k0 + BK, lAs[buf ^ 1] + dstOff);
            stage16(gB + k0 + BK, lBs[buf ^ 1] + dstOff);
        }

        // A fragments: 4 m-tiles for this wave's 64-row strip
        ABfrag a[4];
        #pragma unroll
        for (int mt = 0; mt < 4; ++mt) {
            const unsigned int* src = lAs[buf] + (wm * 64 + mt * 16 + nn) * 16;
            #pragma unroll
            for (int v = 0; v < 8; ++v)
                a[mt].u[v] = src[(v & 3) + 4 * hi + 8 * (v >> 2)];
        }
        #pragma unroll
        for (int nt = 0; nt < 2; ++nt) {
            ABfrag b;
            const unsigned int* src = lBs[buf] + (wn * 32 + nt * 16 + nn) * 16;
            #pragma unroll
            for (int v = 0; v < 8; ++v)
                b.u[v] = src[(v & 3) + 4 * hi + 8 * (v >> 2)];
            #pragma unroll
            for (int mt = 0; mt < 4; ++mt)
                acc[mt][nt] = wmma_bf16(a[mt], b, acc[mt][nt]);
        }
        buf ^= 1;
    }

    // store: C-layout row = v + 8*hi, col = nn
    #pragma unroll
    for (int mt = 0; mt < 4; ++mt)
        #pragma unroll
        for (int nt = 0; nt < 2; ++nt)
            #pragma unroll
            for (int v = 0; v < 8; ++v)
                C[(size_t)(bm + wm * 64 + mt * 16 + v + 8 * hi) * N
                  + bn + wn * 32 + nt * 16 + nn] = acc[mt][nt][v];
}

// ---------------------------------------------------------------------------
// RMSNorm + RoPE + layout change.  One wave per (b,t,h) row; lane j owns head
// dims j and j+32 (exactly the rotary pairing).  Emits:
//   qb,kb : bf16  [B,H,T,64]   (q pre-scaled by 1/sqrt(64))
//   vt    : bf16  [B,H,64,T]   (transposed so PV B-fragments are contiguous)
// ---------------------------------------------------------------------------
__global__ __launch_bounds__(256) void norm_rope(const float* __restrict__ qkv,
                                                 unsigned short* __restrict__ qb,
                                                 unsigned short* __restrict__ kb,
                                                 unsigned short* __restrict__ vt,
                                                 int T) {
    const int wave = threadIdx.x >> 5;
    const int lane = threadIdx.x & 31;
    const int r  = blockIdx.x * 8 + wave;      // (b*T+t)*16 + h
    const int h  = r & 15;
    const int bt = r >> 4;
    const int t  = bt % T;
    const int b  = bt / T;

    const float* row = qkv + (size_t)bt * 3072 + h * 64;
    float q0 = row[lane],        q1 = row[lane + 32];
    float k0 = row[1024 + lane], k1 = row[1024 + lane + 32];
    float v0 = row[2048 + lane], v1 = row[2048 + lane + 32];

    float sq = q0 * q0 + q1 * q1;
    float sk = k0 * k0 + k1 * k1;
    #pragma unroll
    for (int m = 16; m; m >>= 1) {
        sq += __shfl_xor(sq, m, 32);
        sk += __shfl_xor(sk, m, 32);
    }
    const float rq = rsqrtf(sq * (1.0f / 64.0f) + 1e-6f);
    const float rk = rsqrtf(sk * (1.0f / 64.0f) + 1e-6f);
    q0 *= rq; q1 *= rq; k0 *= rk; k1 *= rk;

    // RoPE: inv_freq = 10000^(-2j/64)
    const float invf = __expf(-((float)(2 * lane) * (1.0f / 64.0f)) * 9.210340371976184f);
    const float fr = (float)t * invf;
    const float cs = __cosf(fr), sn = __sinf(fr);
    const float scale = 0.125f;   // 1/sqrt(64) folded into Q

    const float qa = (q0 * cs + q1 * sn) * scale;
    const float qc = (-q0 * sn + q1 * cs) * scale;
    const float ka = k0 * cs + k1 * sn;
    const float kc = -k0 * sn + k1 * cs;

    const size_t bh = (size_t)(b * 16 + h);
    unsigned short* qd = qb + (bh * T + t) * 64;
    qd[lane] = f2bf(qa);  qd[lane + 32] = f2bf(qc);
    unsigned short* kd = kb + (bh * T + t) * 64;
    kd[lane] = f2bf(ka);  kd[lane + 32] = f2bf(kc);
    unsigned short* vd = vt + bh * 64 * (size_t)T;
    vd[(size_t)lane * T + t]        = f2bf(v0);
    vd[(size_t)(lane + 32) * T + t] = f2bf(v1);
}

// ---------------------------------------------------------------------------
// Causal flash attention.  One wave per (b,h,32-query strip): two 16-row
// q-tiles share every K/V fragment.  Per 32-kv step: 8 WMMAs for S, 8 WMMAs
// for P*V against 4 K-frag + 4 V-frag loads.  q0 is 32-aligned so the kv
// loop bound (q0+32) is a multiple of 32 -> no tail tiles; mask handles the
// diagonal.  P converted C-layout f32 -> A-layout bf16 via a 2 KB LDS bounce.
// ---------------------------------------------------------------------------
__global__ __launch_bounds__(32) void flash_attn(const unsigned short* __restrict__ Qb,
                                                 const unsigned short* __restrict__ Kb,
                                                 const unsigned short* __restrict__ Vt,
                                                 float* __restrict__ Y, int T) {
    __shared__ unsigned short lP[2][16 * 32];

    const int nqt = T >> 5;
    const int qt  = blockIdx.x % nqt;
    const int bh  = blockIdx.x / nqt;
    const int h = bh & 15, b = bh >> 4;
    const int q0 = qt * 32;
    const int lane = threadIdx.x;
    const int nn = lane & 15;
    const int hi = lane >> 4;

    // Q A-fragments: two q-tiles, each with K split 0..31 / 32..63
    ABfrag aq[2][2];
    #pragma unroll
    for (int qi = 0; qi < 2; ++qi) {
        const unsigned int* qrow =
            (const unsigned int*)(Qb + ((size_t)bh * T + q0 + qi * 16 + nn) * 64);
        #pragma unroll
        for (int v = 0; v < 8; ++v) {
            const int d = (v & 3) + 4 * hi + 8 * (v >> 2);
            aq[qi][0].u[v] = qrow[d];
            aq[qi][1].u[v] = qrow[16 + d];
        }
    }
    const unsigned short* kpb = Kb + (size_t)bh * T * 64;
    const unsigned short* vpb = Vt + (size_t)bh * 64 * T;

    v8f zero = {};
    v8f o[2][4];
    float mi[2][8], li[2][8];
    #pragma unroll
    for (int qi = 0; qi < 2; ++qi) {
        #pragma unroll
        for (int i = 0; i < 4; ++i) o[qi][i] = zero;
        #pragma unroll
        for (int v = 0; v < 8; ++v) { mi[qi][v] = -1e30f; li[qi][v] = 0.0f; }
    }

    const int kvend = q0 + 32;          // multiple of 32
    for (int j0 = 0; j0 < kvend; j0 += 32) {
        if (j0 + 32 < kvend)            // prefetch next K block (WGP scope)
            __builtin_prefetch(kpb + (size_t)(j0 + 32 + nn) * 64, 0, 3);

        // K fragments for both 16-kv tiles (shared by both q-tiles)
        ABfrag bk[2][2];
        #pragma unroll
        for (int t = 0; t < 2; ++t) {
            const unsigned int* kr =
                (const unsigned int*)(kpb + (size_t)(j0 + t * 16 + nn) * 64);
            #pragma unroll
            for (int v = 0; v < 8; ++v) {
                bk[t][0].u[v] = kr[(v & 3) + 4 * hi + 8 * (v >> 2)];
                bk[t][1].u[v] = kr[16 + (v & 3) + 4 * hi + 8 * (v >> 2)];
            }
        }
        // S = Q K^T : 8 WMMAs
        v8f s[2][2];
        #pragma unroll
        for (int qi = 0; qi < 2; ++qi)
            #pragma unroll
            for (int t = 0; t < 2; ++t) {
                v8f c = wmma_bf16(aq[qi][0], bk[t][0], zero);
                s[qi][t] = wmma_bf16(aq[qi][1], bk[t][1], c);
            }

        // causal mask + online softmax per q-tile
        #pragma unroll
        for (int qi = 0; qi < 2; ++qi) {
            #pragma unroll
            for (int v = 0; v < 8; ++v) {
                const int qg = q0 + qi * 16 + v + 8 * hi;
                if (j0 + nn > qg)      s[qi][0][v] = -1e30f;
                if (j0 + 16 + nn > qg) s[qi][1][v] = -1e30f;
                float rm = fmaxf(s[qi][0][v], s[qi][1][v]);
                #pragma unroll
                for (int m = 1; m < 16; m <<= 1) rm = fmaxf(rm, __shfl_xor(rm, m, 32));
                const float mn = fmaxf(mi[qi][v], rm);
                const float al = __expf(mi[qi][v] - mn);
                mi[qi][v] = mn;
                s[qi][0][v] = __expf(s[qi][0][v] - mn);
                s[qi][1][v] = __expf(s[qi][1][v] - mn);
                float rs = s[qi][0][v] + s[qi][1][v];
                #pragma unroll
                for (int m = 1; m < 16; m <<= 1) rs += __shfl_xor(rs, m, 32);
                li[qi][v] = li[qi][v] * al + rs;
                #pragma unroll
                for (int dt = 0; dt < 4; ++dt) o[qi][dt][v] *= al;
            }
        }

        // P tiles: C-layout f32 -> row-major bf16 in LDS -> A-layout fragments
        __syncthreads();
        #pragma unroll
        for (int qi = 0; qi < 2; ++qi)
            #pragma unroll
            for (int v = 0; v < 8; ++v) {
                lP[qi][(v + 8 * hi) * 32 + nn]      = f2bf(s[qi][0][v]);
                lP[qi][(v + 8 * hi) * 32 + 16 + nn] = f2bf(s[qi][1][v]);
            }
        __syncthreads();
        ABfrag ap[2];
        #pragma unroll
        for (int qi = 0; qi < 2; ++qi) {
            const unsigned int* lpd = (const unsigned int*)lP[qi];
            #pragma unroll
            for (int v = 0; v < 8; ++v)
                ap[qi].u[v] = lpd[nn * 16 + (v & 3) + 4 * hi + 8 * (v >> 2)];
        }
        // O += P * V : V fragments shared by both q-tiles (8 WMMAs)
        #pragma unroll
        for (int dt = 0; dt < 4; ++dt) {
            const unsigned int* vr =
                (const unsigned int*)(vpb + (size_t)(dt * 16 + nn) * T) + (j0 >> 1);
            ABfrag bv;
            #pragma unroll
            for (int v = 0; v < 8; ++v) bv.u[v] = vr[(v & 3) + 4 * hi + 8 * (v >> 2)];
            #pragma unroll
            for (int qi = 0; qi < 2; ++qi)
                o[qi][dt] = wmma_bf16(ap[qi], bv, o[qi][dt]);
        }
        __syncthreads();
    }

    // normalize and store to Y [B*T, 1024] f32
    #pragma unroll
    for (int qi = 0; qi < 2; ++qi)
        #pragma unroll
        for (int dt = 0; dt < 4; ++dt)
            #pragma unroll
            for (int v = 0; v < 8; ++v) {
                const int qg = q0 + qi * 16 + v + 8 * hi;
                Y[((size_t)b * T + qg) * 1024 + h * 64 + dt * 16 + nn] =
                    o[qi][dt][v] / li[qi][v];
            }
}

// ---------------------------------------------------------------------------
// Launcher
// ---------------------------------------------------------------------------
extern "C" void kernel_launch(void* const* d_in, const int* in_sizes, int n_in,
                              void* d_out, int out_size, void* d_ws, size_t ws_size,
                              hipStream_t stream) {
    const float* x     = (const float*)d_in[0];   // [B,T,2048]
    const float* w_qkv = (const float*)d_in[1];   // [3072,2048]
    const float* w_out = (const float*)d_in[2];   // [2048,1024]
    float* out = (float*)d_out;                   // [B,T,2048]

    const int B = 4, T = 2048, H = 16;
    const int Cdim = 2048, L = 1024, F = 3072;
    const int M = B * T;                          // 8192

    char* ws = (char*)d_ws;
    float* qkv = (float*)ws;                                   // M*F f32 (96 MB)
    float* Y   = (float*)ws;                                   // M*L f32, aliases qkv after consumed
    unsigned short* qb = (unsigned short*)(ws + (size_t)M * F * 4);
    unsigned short* kb = qb + (size_t)B * H * T * 64;
    unsigned short* vt = kb + (size_t)B * H * T * 64;

    // 1) qkv = x @ w_qkv^T
    gemm_bf16<<<dim3(F / BN, M / BM), 256, 0, stream>>>(x, w_qkv, qkv, M, F, Cdim);
    // 2) RMSNorm + RoPE + bf16 relayout
    norm_rope<<<(B * T * H) / 8, 256, 0, stream>>>(qkv, qb, kb, vt, T);
    // 3) causal flash attention -> Y f32 [M, 1024]
    flash_attn<<<B * H * (T / 32), 32, 0, stream>>>(qb, kb, vt, Y, T);
    // 4) out = Y @ w_out^T
    gemm_bf16<<<dim3(Cdim / BN, M / BM), 256, 0, stream>>>(Y, w_out, out, M, Cdim, L);
}
